// AttentionBlock_75600014344433
// MI455X (gfx1250) — compile-verified, matrix operand order checked by reading, not run on hardware
//
#include <hip/hip_runtime.h>
#include <hip/hip_bf16.h>

typedef __bf16 bf16;
typedef __attribute__((ext_vector_type(16))) __bf16 v16bf;
typedef __attribute__((ext_vector_type(8)))  __bf16 v8bf;
typedef __attribute__((ext_vector_type(8)))  float   v8f;
typedef __attribute__((ext_vector_type(4)))  int     v4i;

union V16 { v16bf v; v8bf h[2]; };

__device__ __forceinline__ v8f wmma_bf16(v16bf a, v16bf b, v8f c) {
  return __builtin_amdgcn_wmma_f32_16x16x32_bf16(
      /*neg_a=*/false, a, /*neg_b=*/false, b,
      /*c_mod=*/(short)0, c, /*reuse_a=*/false, /*reuse_b=*/false);
}

#if __has_builtin(__builtin_amdgcn_global_load_async_to_lds_b128) && \
    __has_builtin(__builtin_amdgcn_s_wait_asynccnt)
#define HAVE_ASYNC_LDS 1
typedef __attribute__((address_space(1))) v4i* gptr_v4i;
typedef __attribute__((address_space(3))) v4i* sptr_v4i;
#define ASYNC_CP16(dst, src)                                        \
  __builtin_amdgcn_global_load_async_to_lds_b128(                   \
      (gptr_v4i)(v4i*)(src), (sptr_v4i)(v4i*)(dst), 0, 0)
#endif

#define N_TOK   2048
#define HIDDEN  2880
#define QKV_N   5120
#define Q_DIM   4096
#define KV_DIM  512
#define WINDOW  128
#define LDB_PAD 2944      // 23*128: padded out_w leading dim
#define NEG_BIG (-3.0e38f)

// ---------------------------------------------------------------- RMSNorm
__global__ __launch_bounds__(256)
void rmsnorm_kernel(const float* __restrict__ x, const float* __restrict__ w,
                    bf16* __restrict__ t) {
  int row = blockIdx.x;
  const float* xr = x + (size_t)row * HIDDEN;
  __shared__ float red[256];
  float s = 0.0f;
  for (int i = threadIdx.x; i < HIDDEN; i += 256) { float v = xr[i]; s += v * v; }
  red[threadIdx.x] = s;
  __syncthreads();
  for (int o = 128; o > 0; o >>= 1) {
    if (threadIdx.x < o) red[threadIdx.x] += red[threadIdx.x + o];
    __syncthreads();
  }
  float inv = rsqrtf(red[0] / (float)HIDDEN + 1e-5f);
  for (int i = threadIdx.x; i < HIDDEN; i += 256)
    t[(size_t)row * HIDDEN + i] = (bf16)(xr[i] * inv * w[i]);
}

// ---------------------------------------------------------------- f32 -> bf16
__global__ __launch_bounds__(256)
void cvt_kernel(const float* __restrict__ src, bf16* __restrict__ dst, int count) {
  for (int i = blockIdx.x * 256 + threadIdx.x; i < count; i += gridDim.x * 256)
    dst[i] = (bf16)src[i];
}

// f32[rows][cols] -> bf16[rows][ldd], zero-padded columns
__global__ __launch_bounds__(256)
void cvt_pad_kernel(const float* __restrict__ src, bf16* __restrict__ dst,
                    int rows, int cols, int ldd) {
  size_t total = (size_t)rows * ldd;
  for (size_t i = blockIdx.x * 256ULL + threadIdx.x; i < total;
       i += (size_t)gridDim.x * 256ULL) {
    int rr = (int)(i / ldd), cc = (int)(i % ldd);
    dst[i] = (cc < cols) ? (bf16)src[(size_t)rr * cols + cc] : (bf16)0.0f;
  }
}

// ---------------------------------------------------------------- WMMA GEMM
// C[M,N] = A[M,K](bf16) * B[K,*ldb](bf16) (+ residual), f32 out.
// M % 128 == 0, K % 32 == 0, ldb % 128 == 0 (B zero-padded); only C store guarded.
#define BM 128
#define BN 128
#define BK 32
#define LSTR 48   // LDS row stride in halfs (96B, 16B-aligned fragments)

__global__ __launch_bounds__(256)
void gemm_bf16_kernel(const bf16* __restrict__ A, const bf16* __restrict__ B,
                      float* __restrict__ C, const float* __restrict__ residual,
                      int M, int N, int K, int ldb) {
  __shared__ bf16 As[BM * LSTR];
  __shared__ bf16 Bs[BN * LSTR];   // stored transposed: Bs[n][k]
  const int tid  = threadIdx.x;
  const int bm   = blockIdx.y * BM;
  const int bn   = blockIdx.x * BN;
  const int lane = tid & 31;
  const int w    = tid >> 5;
  const int wm   = (w & 1) * 64;
  const int wn   = (w >> 1) * 32;
  const int hlf  = lane >> 4;       // wave half (0/1)
  const int l16  = lane & 15;
  const int c8   = hlf * 8;         // K sub-block per A/B operand layout

  v8f acc[4][2];
  for (int i = 0; i < 4; i++) for (int j = 0; j < 2; j++) acc[i][j] = v8f{};

  const int arow = tid >> 1;
  const int acol = (tid & 1) * 16;
  const int br   = tid >> 3;             // k  in 0..31
  const int bc   = (tid & 7) * 16;       // n0 in 0..112

  for (int k0 = 0; k0 < K; k0 += BK) {
    // ---- stage A tile [BM][BK] (row major) : async DMA to LDS when available
    {
      const bf16* src = A + (size_t)(bm + arow) * K + k0 + acol;
      bf16* dst = &As[arow * LSTR + acol];
#ifdef HAVE_ASYNC_LDS
      ASYNC_CP16(dst,     src);
      ASYNC_CP16(dst + 8, src + 8);
#else
      v8bf x0 = *(const v8bf*)(src);
      v8bf x1 = *(const v8bf*)(src + 8);
      *(v8bf*)(dst)     = x0;
      *(v8bf*)(dst + 8) = x1;
#endif
      if (k0 + BK < K) __builtin_prefetch(src + BK, 0, 1);
    }
    // ---- stage B tile [BK][BN], transposed into Bs[n][k] (no guards: B padded)
    {
      const bf16* src = B + (size_t)(k0 + br) * ldb + bn + bc;
      v8bf x0 = *(const v8bf*)(src);
      v8bf x1 = *(const v8bf*)(src + 8);
      for (int i = 0; i < 8; i++) {
        Bs[(bc + i)     * LSTR + br] = x0[i];
        Bs[(bc + 8 + i) * LSTR + br] = x1[i];
      }
      if (k0 + BK < K) __builtin_prefetch(src + (size_t)BK * ldb, 0, 1);
    }
#ifdef HAVE_ASYNC_LDS
    __builtin_amdgcn_s_wait_asynccnt(0);
#endif
    __syncthreads();

    // ---- fragments (per §7.12.2 16-bit A layout; B mirrored with lane=N)
    V16 af[4], bfr[2];
    for (int mt = 0; mt < 4; mt++) {
      int m = wm + mt * 16 + l16;
      af[mt].h[0] = *(const v8bf*)(&As[m * LSTR + c8]);
      af[mt].h[1] = *(const v8bf*)(&As[m * LSTR + c8 + 16]);
    }
    for (int nt = 0; nt < 2; nt++) {
      int n = wn + nt * 16 + l16;
      bfr[nt].h[0] = *(const v8bf*)(&Bs[n * LSTR + c8]);
      bfr[nt].h[1] = *(const v8bf*)(&Bs[n * LSTR + c8 + 16]);
    }
    for (int mt = 0; mt < 4; mt++)
      for (int nt = 0; nt < 2; nt++)
        acc[mt][nt] = wmma_bf16(af[mt].v, bfr[nt].v, acc[mt][nt]);
    __syncthreads();
  }

  // ---- epilogue (C layout: vgpr r -> row r + 8*half, lane%16 -> col)
  for (int mt = 0; mt < 4; mt++) {
    for (int nt = 0; nt < 2; nt++) {
      int col = bn + wn + nt * 16 + l16;
      if (col >= N) continue;
      for (int r = 0; r < 8; r++) {
        int row = bm + wm + mt * 16 + r + hlf * 8;
        size_t idx = (size_t)row * N + col;
        float v = acc[mt][nt][r];
        if (residual) v += residual[idx];
        C[idx] = v;
      }
    }
  }
}

// ---------------------------------------------------------------- YaRN RoPE + split
__global__ __launch_bounds__(256)
void rope_kernel(const float* __restrict__ qkv, bf16* __restrict__ qo,
                 bf16* __restrict__ ko, bf16* __restrict__ vo) {
  int tok = blockIdx.x;
  __shared__ float cs[32], sn[32];
  if (threadIdx.x < 32) {
    int d = threadIdx.x;
    const float PI2 = 6.28318530717958647692f;
    float freq   = __powf(150000.0f, (float)d / 32.0f);
    float conc   = 0.1f * __logf(32.0f) + 1.0f;
    float lb     = __logf(150000.0f);
    float low    = 32.0f * __logf(4096.0f / (32.0f * PI2)) / lb;
    float high   = 32.0f * __logf(4096.0f / (1.0f * PI2)) / lb;
    float interp = 1.0f / (32.0f * freq);
    float extrap = 1.0f / freq;
    float ramp   = ((float)d - low) / (high - low);
    ramp = fminf(fmaxf(ramp, 0.0f), 1.0f);
    float mask   = 1.0f - ramp;
    float invf   = interp * (1.0f - mask) + extrap * mask;
    float th     = (float)tok * invf;
    cs[d] = cosf(th) * conc;
    sn[d] = sinf(th) * conc;
  }
  __syncthreads();
  const float* base = qkv + (size_t)tok * QKV_N;
  // q: 64 heads * 32 pairs
  for (int p = threadIdx.x; p < 2048; p += 256) {
    int hh = p >> 5, d = p & 31;
    float x1 = base[hh * 64 + d], x2 = base[hh * 64 + d + 32];
    qo[(size_t)tok * Q_DIM + hh * 64 + d]      = (bf16)(x1 * cs[d] - x2 * sn[d]);
    qo[(size_t)tok * Q_DIM + hh * 64 + d + 32] = (bf16)(x2 * cs[d] + x1 * sn[d]);
  }
  // k: 8 heads * 32 pairs
  for (int p = threadIdx.x; p < 256; p += 256) {
    int hh = p >> 5, d = p & 31;
    const float* kb = base + Q_DIM;
    float x1 = kb[hh * 64 + d], x2 = kb[hh * 64 + d + 32];
    ko[(size_t)tok * KV_DIM + hh * 64 + d]      = (bf16)(x1 * cs[d] - x2 * sn[d]);
    ko[(size_t)tok * KV_DIM + hh * 64 + d + 32] = (bf16)(x2 * cs[d] + x1 * sn[d]);
  }
  // v: plain convert
  for (int i = threadIdx.x; i < 512; i += 256)
    vo[(size_t)tok * KV_DIM + i] = (bf16)base[Q_DIM + KV_DIM + i];
}

// ---------------------------------------------------------------- attention
// 1 wave per (head, 16-query tile). Sliding-window flash attention with sink.
__global__ __launch_bounds__(32)
void attn_kernel(const bf16* __restrict__ q, const bf16* __restrict__ k,
                 const bf16* __restrict__ v, const float* __restrict__ sinks,
                 bf16* __restrict__ o) {
  const int nqt = N_TOK / 16;                 // 128 query tiles
  const int h   = blockIdx.x / nqt;           // head 0..63
  const int qi  = blockIdx.x % nqt;
  const int kh  = h >> 3;                     // kv head
  const int lane = threadIdx.x;
  const int hlf  = lane >> 4, l16 = lane & 15, c8 = hlf * 8;
  const int qs   = qi * 16;

  __shared__ bf16 pt[16 * 24];    // P tile, A layout staging (stride 24 halfs)
  __shared__ bf16 vt[64 * LSTR];  // V^T tile, B layout staging (k 16..31 stay 0)

  for (int i = lane; i < 64 * LSTR; i += 32) vt[i] = (bf16)0.0f;
  __syncthreads();

  // Q fragments (two K=32 chunks over head dim 64)
  const bf16* qrow = q + (size_t)(qs + l16) * Q_DIM + h * 64;
  V16 qa0, qa1;
  qa0.h[0] = *(const v8bf*)(qrow + c8);
  qa0.h[1] = *(const v8bf*)(qrow + c8 + 16);
  qa1.h[0] = *(const v8bf*)(qrow + 32 + c8);
  qa1.h[1] = *(const v8bf*)(qrow + 32 + c8 + 16);

  const float sink = sinks[h];
  float m_[8], l_[8];
  for (int r = 0; r < 8; r++) { m_[r] = sink; l_[r] = 1.0f; }
  v8f acc[4];
  for (int d = 0; d < 4; d++) acc[d] = v8f{};
  const float sm = 0.125f;  // 1/sqrt(64)

  int t0 = qi - 8; if (t0 < 0) t0 = 0;
  for (int t = t0; t <= qi; t++) {
    const int ks = t * 16;
    // K fragment: B operand, lane = key
    const bf16* krow = k + (size_t)(ks + l16) * KV_DIM + kh * 64;
    V16 kb0, kb1;
    kb0.h[0] = *(const v8bf*)(krow + c8);
    kb0.h[1] = *(const v8bf*)(krow + c8 + 16);
    kb1.h[0] = *(const v8bf*)(krow + 32 + c8);
    kb1.h[1] = *(const v8bf*)(krow + 32 + c8 + 16);

    v8f s = v8f{};
    s = wmma_bf16(qa0.v, kb0.v, s);
    s = wmma_bf16(qa1.v, kb1.v, s);

    // scale + causal/sliding-window mask (row = query, col = key)
    float sv[8];
    const int col = ks + l16;
    for (int r = 0; r < 8; r++) {
      int row = qs + r + hlf * 8;
      float x = s[r] * sm;
      if (col > row || row - col >= WINDOW) x = NEG_BIG;
      sv[r] = x;
    }
    // row max across the 16-lane group
    float rmax[8];
    for (int r = 0; r < 8; r++) {
      float x = sv[r];
      for (int off = 1; off < 16; off <<= 1)
        x = fmaxf(x, __shfl_xor(x, off, 32));
      rmax[r] = x;
    }
    // online softmax update
    float p[8], corr[8];
    for (int r = 0; r < 8; r++) {
      float mnew = fmaxf(m_[r], rmax[r]);
      corr[r] = __expf(m_[r] - mnew);
      m_[r]   = mnew;
      p[r]    = __expf(sv[r] - mnew);
    }
    for (int r = 0; r < 8; r++) {
      float x = p[r];
      for (int off = 1; off < 16; off <<= 1)
        x += __shfl_xor(x, off, 32);
      l_[r] = l_[r] * corr[r] + x;
    }
    for (int d = 0; d < 4; d++)
      for (int r = 0; r < 8; r++) acc[d][r] *= corr[r];

    // stage P (bf16) into A layout, stage V^T into B layout
    for (int r = 0; r < 8; r++)
      pt[(r + 8 * hlf) * 24 + l16] = (bf16)p[r];
    {
      int key = lane >> 1;
      int db  = (lane & 1) * 32;
      const bf16* vrow = v + (size_t)(ks + key) * KV_DIM + kh * 64 + db;
      v8bf v0 = *(const v8bf*)(vrow);
      v8bf v1 = *(const v8bf*)(vrow + 8);
      v8bf v2 = *(const v8bf*)(vrow + 16);
      v8bf v3 = *(const v8bf*)(vrow + 24);
      for (int i = 0; i < 8; i++) {
        vt[(db + i)      * LSTR + key] = v0[i];
        vt[(db + 8 + i)  * LSTR + key] = v1[i];
        vt[(db + 16 + i) * LSTR + key] = v2[i];
        vt[(db + 24 + i) * LSTR + key] = v3[i];
      }
    }
    __syncthreads();  // single-wave WG: ordering fence

    V16 pa;
    pa.h[0] = *(const v8bf*)(&pt[l16 * 24 + c8]);
    pa.h[1] = v8bf{};           // keys 16..31 are zero padding
    for (int d = 0; d < 4; d++) {
      V16 vb;
      int n = d * 16 + l16;
      vb.h[0] = *(const v8bf*)(&vt[n * LSTR + c8]);
      vb.h[1] = *(const v8bf*)(&vt[n * LSTR + c8 + 16]);  // zeros
      acc[d] = wmma_bf16(pa.v, vb.v, acc[d]);
    }
    __syncthreads();
  }

  // epilogue: O = acc / l, write bf16 [tok][head*64+dim]
  for (int d = 0; d < 4; d++) {
    int col = h * 64 + d * 16 + l16;
    for (int r = 0; r < 8; r++) {
      int row = qs + r + 8 * hlf;
      o[(size_t)row * Q_DIM + col] = (bf16)(acc[d][r] / l_[r]);
    }
  }
}

// ---------------------------------------------------------------- launch
extern "C" void kernel_launch(void* const* d_in, const int* in_sizes, int n_in,
                              void* d_out, int out_size, void* d_ws, size_t ws_size,
                              hipStream_t stream) {
  const float* x      = (const float*)d_in[0];
  const float* norm_w = (const float*)d_in[1];
  const float* qkv_w  = (const float*)d_in[2];
  const float* out_w  = (const float*)d_in[3];
  const float* sinks  = (const float*)d_in[4];
  float* out = (float*)d_out;

  char* ws = (char*)d_ws;
  size_t off = 0;
  auto alloc = [&](size_t bytes) -> void* {
    void* p = ws + off;
    off += (bytes + 255) & ~(size_t)255;
    return p;
  };
  bf16*  t     = (bf16*)alloc((size_t)N_TOK * HIDDEN * 2);
  bf16*  wqkv  = (bf16*)alloc((size_t)HIDDEN * QKV_N * 2);
  bf16*  wout  = (bf16*)alloc((size_t)Q_DIM * LDB_PAD * 2);
  float* qkv   = (float*)alloc((size_t)N_TOK * QKV_N * 4);
  bf16*  qb    = (bf16*)alloc((size_t)N_TOK * Q_DIM * 2);
  bf16*  kb    = (bf16*)alloc((size_t)N_TOK * KV_DIM * 2);
  bf16*  vb    = (bf16*)alloc((size_t)N_TOK * KV_DIM * 2);
  bf16*  attnb = (bf16*)alloc((size_t)N_TOK * Q_DIM * 2);

  // 1. RMSNorm -> bf16
  rmsnorm_kernel<<<N_TOK, 256, 0, stream>>>(x, norm_w, t);

  // 2. weight conversion (out_w padded to ldb=2944 with zeros)
  {
    int c1 = HIDDEN * QKV_N;
    cvt_kernel<<<(c1 + 255) / 256, 256, 0, stream>>>(qkv_w, wqkv, c1);
    size_t c2 = (size_t)Q_DIM * LDB_PAD;
    cvt_pad_kernel<<<(int)((c2 + 255) / 256), 256, 0, stream>>>(
        out_w, wout, Q_DIM, HIDDEN, LDB_PAD);
  }

  // 3. QKV GEMM: [2048,2880] x [2880,5120] -> f32
  gemm_bf16_kernel<<<dim3(QKV_N / BN, N_TOK / BM), 256, 0, stream>>>(
      t, wqkv, qkv, nullptr, N_TOK, QKV_N, HIDDEN, QKV_N);

  // 4. RoPE + split
  rope_kernel<<<N_TOK, 256, 0, stream>>>(qkv, qb, kb, vb);

  // 5. sliding-window attention with sinks
  attn_kernel<<<64 * (N_TOK / 16), 32, 0, stream>>>(qb, kb, vb, sinks, attnb);

  // 6. out GEMM + residual: [2048,4096] x [4096,2880(pad 2944)] + x -> f32 out
  gemm_bf16_kernel<<<dim3(LDB_PAD / BN, N_TOK / BM), 256, 0, stream>>>(
      attnb, wout, out, x, N_TOK, HIDDEN, Q_DIM, LDB_PAD);
}
